// conv3d_45603962749212
// MI455X (gfx1250) — matrix operand order repaired and checked by reading, not compiled, not run on hardware
//
#include <hip/hip_runtime.h>

#define N_VOX   131072
#define KVOL    27
#define PAIRS   65536
#define CIN     64
#define COUT    64
#define TILE_P  256
#define AS_STRIDE 72   // halves per A row (64 + pad), conflict-free fragment reads
#define BS_STRIDE 72   // halves per B^T row

typedef __attribute__((ext_vector_type(16))) __bf16 v16bf;
typedef __attribute__((ext_vector_type(8)))  float  v8f;

__device__ __forceinline__ unsigned int pack_bf16(float a, float b) {
    unsigned int ua = __builtin_bit_cast(unsigned int, a);
    unsigned int ub = __builtin_bit_cast(unsigned int, b);
    return (ua >> 16) | (ub & 0xFFFF0000u);
}
__device__ __forceinline__ unsigned short bf16_of(float f) {
    return (unsigned short)(__builtin_bit_cast(unsigned int, f) >> 16);
}

union AFrag { unsigned int u[8]; v16bf v; };

// out[i] = bias[i % 64]  (seed for the scatter-add; also applies +bias exactly once)
__global__ void sparseconv_init_kernel(const float* __restrict__ bias,
                                       float* __restrict__ out, int total) {
    int i = blockIdx.x * blockDim.x + threadIdx.x;
    if (i < total) out[i] = bias[i & (COUT - 1)];
}

__global__ __launch_bounds__(256, 2)
void sparseconv_wmma_kernel(const float* __restrict__ in_feats,
                            const float* __restrict__ kernel_w,
                            const int*   __restrict__ imap,
                            const int*   __restrict__ omap,
                            float*       __restrict__ out) {
    __shared__ unsigned short As[TILE_P * AS_STRIDE]; // gathered rows, bf16 (36 KB)
    __shared__ unsigned short Bs[COUT * BS_STRIDE];   // B^T: Bs[n][c] = W[c][n] (9 KB)
    __shared__ int omap_s[TILE_P];

    const int tid = threadIdx.x;
    const int k  = blockIdx.y;                 // kernel offset 0..26
    const int p0 = blockIdx.x * TILE_P;        // pair-tile base

    // ---- gather phase: 1 thread per row, 64 cols, fp32 -> bf16 -> LDS ----
    {
        const int row  = tid;                  // 0..255
        const int gidx = imap[k * PAIRS + p0 + row];
        const float4* src = (const float4*)(in_feats + (long)gidx * CIN);
        unsigned int* dstA = (unsigned int*)(As + row * AS_STRIDE);
        #pragma unroll
        for (int j = 0; j < 16; ++j) {
            float4 v = src[j];
            dstA[2 * j + 0] = pack_bf16(v.x, v.y);
            dstA[2 * j + 1] = pack_bf16(v.z, v.w);
        }
        omap_s[tid] = omap[k * PAIRS + p0 + tid];
    }

    // ---- weight phase: transpose W[k] (CIN x COUT) into Bs[n][c], bf16 ----
    {
        const float* wk = kernel_w + k * (CIN * COUT);
        #pragma unroll
        for (int idx = tid; idx < CIN * COUT; idx += 256) {
            int c = idx >> 6;          // CIN index
            int n = idx & 63;          // COUT index
            Bs[n * BS_STRIDE + c] = bf16_of(wk[idx]);
        }
    }
    __syncthreads();

    // ---- compute: wave owns M-tiles {wave, wave+8}; 4 N-tiles x 2 M x 2 K = 16 WMMA ----
    const int wave = tid >> 5;         // 0..7
    const int lane = tid & 31;
    const int half = lane >> 4;        // 0: lanes 0-15, 1: lanes 16-31
    const int l    = lane & 15;

    // A fragments (ISA 16-bit A 16x32 layout): [mtile][kstep], hoisted out of nt loop
    AFrag a[2][2];
    #pragma unroll
    for (int t = 0; t < 2; ++t) {
        const unsigned short* arow = As + ((wave + t * 8) * 16 + l) * AS_STRIDE;
        #pragma unroll
        for (int j = 0; j < 4; ++j) {
            a[t][0].u[j]     = *(const unsigned int*)(arow +  0 + half * 8 + j * 2);
            a[t][0].u[4 + j] = *(const unsigned int*)(arow + 16 + half * 8 + j * 2);
            a[t][1].u[j]     = *(const unsigned int*)(arow + 32 + half * 8 + j * 2);
            a[t][1].u[4 + j] = *(const unsigned int*)(arow + 48 + half * 8 + j * 2);
        }
    }

    #pragma unroll
    for (int nt = 0; nt < 4; ++nt) {
        // B fragments (ISA 16-bit B 32x16 layout): lane holds column n = nt*16 + l.
        // Loaded once per N-tile, reused by both M-tiles.
        const unsigned short* bcol = Bs + (nt * 16 + l) * BS_STRIDE;
        AFrag b0, b1;
        #pragma unroll
        for (int j = 0; j < 4; ++j) {
            b0.u[j]     = *(const unsigned int*)(bcol +  0 + half * 8 + j * 2);
            b0.u[4 + j] = *(const unsigned int*)(bcol + 16 + half * 8 + j * 2);
            b1.u[j]     = *(const unsigned int*)(bcol + 32 + half * 8 + j * 2);
            b1.u[4 + j] = *(const unsigned int*)(bcol + 48 + half * 8 + j * 2);
        }

        #pragma unroll
        for (int t = 0; t < 2; ++t) {
            v8f acc = {};
            acc = __builtin_amdgcn_wmma_f32_16x16x32_bf16(
                      false, a[t][0].v, false, b0.v, (short)0, acc, false, false);
            acc = __builtin_amdgcn_wmma_f32_16x16x32_bf16(
                      false, a[t][1].v, false, b1.v, (short)0, acc, false, false);

            // ---- scatter-add: D VGPR r -> (M, N = nt*16 + l) ----
            #pragma unroll
            for (int r = 0; r < 8; ++r) {
                int m = (wave + t * 8) * 16 + half * 8 + r;
                int orow = omap_s[m];
                unsafeAtomicAdd(out + (long)orow * COUT + nt * 16 + l, acc[r]);
            }
        }
    }
}

extern "C" void kernel_launch(void* const* d_in, const int* in_sizes, int n_in,
                              void* d_out, int out_size, void* d_ws, size_t ws_size,
                              hipStream_t stream) {
    const float* in_feats = (const float*)d_in[0];
    const float* kernel_w = (const float*)d_in[1];
    const float* bias     = (const float*)d_in[2];
    const int*   imap     = (const int*)d_in[3];
    const int*   omap     = (const int*)d_in[4];
    float*       out      = (float*)d_out;

    // 1) seed output with bias (also clears poison)
    int initBlocks = (out_size + 255) / 256;
    sparseconv_init_kernel<<<initBlocks, 256, 0, stream>>>(bias, out, out_size);

    // 2) gather -> bf16 WMMA GEMM -> f32 atomic scatter-add
    dim3 grid(PAIRS / TILE_P, KVOL);   // (256, 27)
    sparseconv_wmma_kernel<<<grid, 256, 0, stream>>>(in_feats, kernel_w, imap, omap, out);
}
